// CNN_FS_MGDA_SEGREL_29283087024820
// MI455X (gfx1250) — matrix operand order repaired and checked by reading, not compiled, not run on hardware
//
#include <hip/hip_runtime.h>
#include <hip/hip_bf16.h>
#include <math.h>
#include <cstdint>

// ---------------------------------------------------------------------------
// Problem constants (from reference): S=256, Q=4096, T=8, D=1024, K=64 classes
// ---------------------------------------------------------------------------
#define D_DIM      1024
#define Q_NUM      4096
#define S_NUM      256
#define T_NUM      8
#define K_CLS      64
#define QK         (Q_NUM * K_CLS)          // 262144 per output tensor

#define A_ROW_PAD  1032                     // 1024 + 8 bf16 -> 2064B row stride
                                            // (516 dwords % 64 banks = 4 -> conflict-free)
#define A_TILE_BYTES (64 * A_ROW_PAD * 2)   // 132096
#define EPI_BYTES    (4 * 64 * 65 * 4)      // 66560 (per-wave 64x65 f32)
#define SMEM_BYTES   (A_TILE_BYTES + EPI_BYTES)

typedef __attribute__((ext_vector_type(16))) __bf16 v16bf;
typedef __attribute__((ext_vector_type(8)))  float  v8f;
typedef int vint4 __attribute__((vector_size(16)));

// ---------------------------------------------------------------------------
// Async global->LDS copy (CDNA5 GLOBAL_LOAD_ASYNC_TO_LDS_B128, ASYNCcnt).
// Builtin signature (from hipcc diagnostic): (v4i AS1*, v4i AS3*, imm, imm).
// Guarded: falls back to load+ds_store if the builtin is unavailable
// (also keeps the host compilation pass happy).
// ---------------------------------------------------------------------------
#if defined(__AMDGCN__) && __has_builtin(__builtin_amdgcn_global_load_async_to_lds_b128)
#define HAVE_ASYNC_LDS 1
#else
#define HAVE_ASYNC_LDS 0
#endif

#define AS1 __attribute__((address_space(1)))
#define AS3 __attribute__((address_space(3)))

__device__ __forceinline__ void async_copy16(const __bf16* gp, __bf16* lp) {
#if HAVE_ASYNC_LDS
    __builtin_amdgcn_global_load_async_to_lds_b128(
        (AS1 vint4*)(unsigned long long)(uintptr_t)gp,
        (AS3 vint4*)(unsigned)(uintptr_t)lp, 0, 0);
#else
    *(uint4*)lp = *(const uint4*)gp;
#endif
}

__device__ __forceinline__ void wait_async0() {
#if HAVE_ASYNC_LDS
#if __has_builtin(__builtin_amdgcn_s_wait_asynccnt)
    __builtin_amdgcn_s_wait_asynccnt(0);
#else
    asm volatile("s_wait_asynccnt 0" ::: "memory");
#endif
#endif
}

// ---------------------------------------------------------------------------
// Fragment loaders matching CDNA5 16-bit WMMA VGPR layouts (wave32).
// A (16x32): lane l -> row M = l&15, half h = l>>4,
//   elems 0..7 = K (h*8..h*8+7), elems 8..15 = K (16+h*8..).
// B (32x16, B[k][n] = rowmajor[n][k]): lane l -> col N = l&15,
//   elems 0..7 = K (h*8..h*8+7), elems 8..15 = K (16+h*8..).
// ---------------------------------------------------------------------------
__device__ __forceinline__ v16bf load_afrag_lds(const __bf16* smA,
                                                int mloc, int k0, int lane) {
    const int h = lane >> 4;
    const __bf16* p = smA + mloc * A_ROW_PAD + k0 + h * 8;
    union { uint4 u[2]; v16bf v; } t;
    t.u[0] = *(const uint4*)(p);
    t.u[1] = *(const uint4*)(p + 16);
    return t.v;
}

__device__ __forceinline__ v16bf load_bfrag(const __bf16* __restrict__ B,
                                            int colBase, int k0, int lane) {
    const int n = lane & 15;
    const int h = lane >> 4;
    const __bf16* p = B + (size_t)(colBase + n) * D_DIM + k0 + h * 8;
    union { uint4 u[2]; v16bf v; } t;
    t.u[0] = *(const uint4*)(p);
    t.u[1] = *(const uint4*)(p + 16);
    return t.v;
}

// Fill the 64-row x 1024-K A tile (padded rows) with async LDS loads.
// 8192 x 16B chunks, 128 threads, 64 per thread. ASYNCcnt self-throttles.
__device__ __forceinline__ void fill_atile(__bf16* smA,
                                           const __bf16* __restrict__ A,
                                           int mBase, int tid) {
#pragma unroll 4
    for (int it = 0; it < 64; ++it) {
        const int chunk = it * 128 + tid;
        const int row = chunk >> 7;          // 128 chunks of 8 elems per row
        const int c8  = (chunk & 127) * 8;
        const __bf16* gp = A + (size_t)(mBase + row) * D_DIM + c8;
        __bf16* lp = smA + row * A_ROW_PAD + c8;
        async_copy16(gp, lp);
    }
    wait_async0();
}

// One wave: 64(M) x 64(N) tile as 4x4 WMMA accumulators, A from LDS,
// B from global (L2-resident); result staged to wave-private LDS epi[64][65].
__device__ __forceinline__ void gemm_tile_64x64(const __bf16* smA,
                                                const __bf16* __restrict__ B,
                                                int colBase, int lane,
                                                float* epi) {
    v8f acc[4][4];
    const v8f vz = {0.f, 0.f, 0.f, 0.f, 0.f, 0.f, 0.f, 0.f};
#pragma unroll
    for (int i = 0; i < 4; ++i)
#pragma unroll
        for (int j = 0; j < 4; ++j) acc[i][j] = vz;

    for (int k0 = 0; k0 < D_DIM; k0 += 32) {
        v16bf af[4], bfr[4];
#pragma unroll
        for (int i = 0; i < 4; ++i)
            af[i] = load_afrag_lds(smA, i * 16 + (lane & 15), k0, lane);
#pragma unroll
        for (int j = 0; j < 4; ++j)
            bfr[j] = load_bfrag(B, colBase + j * 16, k0, lane);
#pragma unroll
        for (int i = 0; i < 4; ++i)
#pragma unroll
            for (int j = 0; j < 4; ++j)
                acc[i][j] = __builtin_amdgcn_wmma_f32_16x16x32_bf16(
                    false, af[i], false, bfr[j], (short)0, acc[i][j],
                    false, false);
    }
    // C/D layout: element (m,n): m = v + 8*(lane>>4), n = lane&15
    const int mh = (lane >> 4) << 3;
    const int nn = lane & 15;
#pragma unroll
    for (int i = 0; i < 4; ++i)
#pragma unroll
        for (int j = 0; j < 4; ++j)
#pragma unroll
            for (int v = 0; v < 8; ++v)
                epi[(i * 16 + mh + v) * 65 + j * 16 + nn] = acc[i][j][v];
}

// ---------------------------------------------------------------------------
// Frame-level GEMM + bidirectional min-sum (Chamfer) epilogue.
// A: [32768,1024] bf16 (q*8+t), B: [512,1024] bf16 (k*8+s), out [4096,64] f32.
// WG = 64 A-rows staged once in LDS; 4 waves x 2 n-iters cover all 512 cols.
// dist = 1 - sim  =>  result = 16 - sum_t max_s sim - sum_s max_t sim.
// ---------------------------------------------------------------------------
__global__ __launch_bounds__(128) void gemm_bimhm_global(
    const __bf16* __restrict__ A, const __bf16* __restrict__ B,
    float* __restrict__ out) {
    extern __shared__ char smem_raw[];
    __bf16* smA = (__bf16*)smem_raw;
    const int tid = threadIdx.x, w = tid >> 5, lane = tid & 31;
    const int mBase = blockIdx.x * 64;

    fill_atile(smA, A, mBase, tid);
    __syncthreads();

    float* epi = (float*)(smem_raw + A_TILE_BYTES) + w * (64 * 65);
    const int qb = blockIdx.x * 8;                     // 64 rows / 8 frames
#pragma unroll
    for (int nit = 0; nit < 2; ++nit) {
        const int colBase = (nit * 4 + w) * 64;
        gemm_tile_64x64(smA, B, colBase, lane, epi);
        __syncthreads();
        const int kb = colBase >> 3;
        for (int p = lane; p < 64; p += 32) {
            const int qi = p >> 3, ki = p & 7;
            float colmax[8];
#pragma unroll
            for (int s = 0; s < 8; ++s) colmax[s] = -1e30f;
            float rowsum = 0.f;
#pragma unroll
            for (int t = 0; t < 8; ++t) {
                const float* rp = epi + (qi * 8 + t) * 65 + ki * 8;
                float rmax = -1e30f;
#pragma unroll
                for (int s = 0; s < 8; ++s) {
                    const float x = rp[s];
                    rmax = fmaxf(rmax, x);
                    colmax[s] = fmaxf(colmax[s], x);
                }
                rowsum += rmax;
            }
            float colsum = 0.f;
#pragma unroll
            for (int s = 0; s < 8; ++s) colsum += colmax[s];
            out[(size_t)(qb + qi) * K_CLS + (kb + ki)] = 16.0f - rowsum - colsum;
        }
        __syncthreads();
    }
}

// ---------------------------------------------------------------------------
// Segment-level GEMM (4-padded segments; pad rows/cols are zero and ignored).
// A: [16384,1024] (q*4+seg), B: [256,1024] (k*4+seg). 4 waves x 1 n-iter.
// ---------------------------------------------------------------------------
__global__ __launch_bounds__(128) void gemm_seg(
    const __bf16* __restrict__ A, const __bf16* __restrict__ B,
    float* __restrict__ out_q2s, float* __restrict__ out_s2q) {
    extern __shared__ char smem_raw[];
    __bf16* smA = (__bf16*)smem_raw;
    const int tid = threadIdx.x, w = tid >> 5, lane = tid & 31;
    const int mBase = blockIdx.x * 64;

    fill_atile(smA, A, mBase, tid);
    __syncthreads();

    float* epi = (float*)(smem_raw + A_TILE_BYTES) + w * (64 * 65);
    const int qb = blockIdx.x * 16;                    // 64 rows / 4 slots
    const int colBase = w * 64;                        // 4 waves cover N=256
    gemm_tile_64x64(smA, B, colBase, lane, epi);
    __syncthreads();
    const int kb = w * 16;
    for (int p = lane; p < 256; p += 32) {
        const int qi = p >> 4, ki = p & 15;
        float colmin[3] = {1e30f, 1e30f, 1e30f};
        float q2s = 0.f;
#pragma unroll
        for (int t = 0; t < 3; ++t) {
            const float* rp = epi + (qi * 4 + t) * 65 + ki * 4;
            float rmin = 1e30f;
#pragma unroll
            for (int s = 0; s < 3; ++s) {
                const float d = 1.0f - rp[s];
                rmin = fminf(rmin, d);
                colmin[s] = fminf(colmin[s], d);
            }
            q2s += rmin;
        }
        const size_t o = (size_t)(qb + qi) * K_CLS + (kb + ki);
        out_q2s[o] = q2s;
        out_s2q[o] = colmin[0] + colmin[1] + colmin[2];
    }
}

// ---------------------------------------------------------------------------
// Preprocessing kernels
// ---------------------------------------------------------------------------
__device__ __forceinline__ float block_reduce_sum_256(float v, float* sbuf) {
    const int tid = threadIdx.x;
    sbuf[tid] = v;
    __syncthreads();
    for (int s = 128; s > 0; s >>= 1) {
        if (tid < s) sbuf[tid] += sbuf[tid + s];
        __syncthreads();
    }
    const float r = sbuf[0];
    __syncthreads();
    return r;
}

__global__ void zero_f32(float* __restrict__ p, int n) {
    const int i = blockIdx.x * blockDim.x + threadIdx.x;
    if (i < n) p[i] = 0.f;
}

// scatter-add support features into per-class sums (class counts drop out
// after L2 normalization, so they are never needed).
__global__ void proto_accum(const float* __restrict__ sup,
                            const int* __restrict__ labels,
                            float* __restrict__ protoSum) {
    const int idx = blockIdx.x * blockDim.x + threadIdx.x;  // S*T*D
    const int s  = idx >> 13;        // / (T*D) = 8192
    const int td = idx & 8191;
    atomicAdd(&protoSum[(size_t)labels[s] * (T_NUM * D_DIM) + td], sup[idx]);
}

// L2-normalize one 1024-wide f32 row -> bf16 row.
__global__ void rownorm_f32_to_bf16(const float* __restrict__ in,
                                    __bf16* __restrict__ out) {
    __shared__ float sbuf[256];
    const float* row = in + (size_t)blockIdx.x * D_DIM;
    float ss = 0.f;
    for (int d = threadIdx.x; d < D_DIM; d += 256) {
        const float x = row[d];
        ss += x * x;
    }
    ss = block_reduce_sum_256(ss, sbuf);
    const float scale = 1.0f / fmaxf(sqrtf(ss), 1e-12f);
    __bf16* orow = out + (size_t)blockIdx.x * D_DIM;
    for (int d = threadIdx.x; d < D_DIM; d += 256)
        orow[d] = (__bf16)(row[d] * scale);
}

// windows (0,4),(2,6),(4,8): start = 2*seg, len 4. Output is 4-slot padded
// [N,4,1024] bf16; slot 3 written as zeros (ignored by GEMM epilogue).
__global__ void segnorm_f32_to_bf16(const float* __restrict__ in /*[N,8,1024]*/,
                                    __bf16* __restrict__ out /*[N,4,1024]*/) {
    __shared__ float sbuf[256];
    const int n = blockIdx.x >> 2, seg = blockIdx.x & 3;
    __bf16* orow = out + (size_t)blockIdx.x * D_DIM;
    if (seg == 3) {
        for (int d = threadIdx.x; d < D_DIM; d += 256) orow[d] = (__bf16)0.f;
        return;
    }
    const float* base = in + (size_t)n * (T_NUM * D_DIM) + (size_t)(2 * seg) * D_DIM;
    float vals[4];
    float ss = 0.f;
#pragma unroll
    for (int c = 0; c < 4; ++c) {
        const int d = threadIdx.x + c * 256;
        const float m = 0.25f * (base[d] + base[d + D_DIM] +
                                 base[d + 2 * D_DIM] + base[d + 3 * D_DIM]);
        vals[c] = m;
        ss += m * m;
    }
    ss = block_reduce_sum_256(ss, sbuf);
    const float scale = 1.0f / fmaxf(sqrtf(ss), 1e-12f);
#pragma unroll
    for (int c = 0; c < 4; ++c)
        orow[threadIdx.x + c * 256] = (__bf16)(vals[c] * scale);
}

// fused = softmax(fusion)[0]*g + [1]*s2q + [2]*q2s, scaled by exp(logit_scale)
__global__ void finalize(const float* __restrict__ gdist,
                         const float* __restrict__ q2s,
                         const float* __restrict__ s2q,
                         const float* __restrict__ logit_scale,
                         const float* __restrict__ fusion_logits,
                         float* __restrict__ out) {
    const int idx = blockIdx.x * blockDim.x + threadIdx.x;  // QK
    const float f0 = fusion_logits[0], f1 = fusion_logits[1], f2 = fusion_logits[2];
    const float m  = fmaxf(f0, fmaxf(f1, f2));
    const float e0 = expf(f0 - m), e1 = expf(f1 - m), e2 = expf(f2 - m);
    const float inv = 1.0f / (e0 + e1 + e2);
    const float ls  = expf(logit_scale[0]);
    const float g = gdist[idx], a = s2q[idx], b = q2s[idx];
    const float fused = (e0 * inv * g + e1 * inv * a + e2 * inv * b) * ls;
    out[idx]          = -fused;
    out[QK + idx]     = -g;
    out[2 * QK + idx] = -a;
    out[3 * QK + idx] = -b;
}

// ---------------------------------------------------------------------------
// Launcher
// ---------------------------------------------------------------------------
extern "C" void kernel_launch(void* const* d_in, const int* in_sizes, int n_in,
                              void* d_out, int out_size, void* d_ws, size_t ws_size,
                              hipStream_t stream) {
    const float* sup    = (const float*)d_in[0];   // [256, 8, 1024]
    const float* tgt    = (const float*)d_in[1];   // [4096, 8, 1024]
    const int*   labels = (const int*)  d_in[2];   // [256]
    const float* lscale = (const float*)d_in[3];   // [1]
    const float* flog   = (const float*)d_in[4];   // [3]
    float*       out    = (float*)d_out;           // 4 * 262144

    // workspace carve (256B aligned slabs)
    char* ws = (char*)d_ws;
    auto take = [&](size_t bytes) {
        char* p = ws;
        ws += (bytes + 255) & ~(size_t)255;
        return p;
    };
    float*  protoSum  = (float*) take((size_t)K_CLS * T_NUM * D_DIM * 4);  //  2 MB
    __bf16* protoN    = (__bf16*)take((size_t)K_CLS * T_NUM * D_DIM * 2);  //  1 MB  [512,1024]
    __bf16* protoSegN = (__bf16*)take((size_t)K_CLS * 4 * D_DIM * 2);      // .5 MB  [256,1024]
    __bf16* targetN   = (__bf16*)take((size_t)Q_NUM * T_NUM * D_DIM * 2);  // 64 MB  [32768,1024]
    __bf16* targetSegN= (__bf16*)take((size_t)Q_NUM * 4 * D_DIM * 2);      // 32 MB  [16384,1024]
    float*  gdist     = (float*) take((size_t)QK * 4);
    float*  q2s       = (float*) take((size_t)QK * 4);
    float*  s2q       = (float*) take((size_t)QK * 4);

    // 1) class-prototype sums
    zero_f32<<<(K_CLS * T_NUM * D_DIM + 255) / 256, 256, 0, stream>>>(
        protoSum, K_CLS * T_NUM * D_DIM);
    proto_accum<<<(S_NUM * T_NUM * D_DIM) / 256, 256, 0, stream>>>(
        sup, labels, protoSum);

    // 2) normalize + convert to bf16 (frame rows and 4-padded segment rows)
    rownorm_f32_to_bf16<<<K_CLS * T_NUM, 256, 0, stream>>>(protoSum, protoN);
    segnorm_f32_to_bf16<<<K_CLS * 4, 256, 0, stream>>>(protoSum, protoSegN);
    rownorm_f32_to_bf16<<<Q_NUM * T_NUM, 256, 0, stream>>>(tgt, targetN);
    segnorm_f32_to_bf16<<<Q_NUM * 4, 256, 0, stream>>>(tgt, targetSegN);

    // 3) WMMA GEMMs (A staged in LDS via async loads) + fused Chamfer epilogues
    gemm_bimhm_global<<<(Q_NUM * T_NUM) / 64, 128, SMEM_BYTES, stream>>>(
        targetN, protoN, gdist);
    gemm_seg<<<(Q_NUM * 4) / 64, 128, SMEM_BYTES, stream>>>(
        targetSegN, protoSegN, q2s, s2q);

    // 4) fusion + outputs
    finalize<<<QK / 256, 256, 0, stream>>>(gdist, q2s, s2q, lscale, flog, out);
}
